// LCABlock_66872640799334
// MI455X (gfx1250) — matrix operand order repaired and checked by reading, not compiled, not run on hardware
//
#include <hip/hip_runtime.h>
#include <hip/hip_bf16.h>

// LCAB (Restormer-style cross channel-attention block) for MI455X / gfx1250.
// Pointwise GEMMs on v_wmma_f32_16x16x32_f16 (wave32 WMMA); activation tiles
// streamed into LDS by the Tensor Data Mover (tensor_load_to_lds + TENSORcnt).
//
// Shapes: B=4, C=64, H=W=256, N=H*W=65536, P=B*N=262144, HID=170.
// Planar f16 intermediates: buf[c*P + p], p = b*65536 + n.
// Input/Output layout (NCHW): off = b*64*65536 + c*65536 + n.

#define DIMC 64
#define NPIX 65536
#define PTOT 262144   // == 1<<18
#define HEADS 8

typedef __attribute__((ext_vector_type(16))) _Float16 v16h;
typedef __attribute__((ext_vector_type(8)))  float    v8f;
typedef __attribute__((ext_vector_type(4)))  unsigned int v4u;
typedef __attribute__((ext_vector_type(8)))  int      v8i;
typedef __attribute__((ext_vector_type(4)))  int      v4i;

union HFrag {
    v16h v;
    _Float16 h[16];
    unsigned int d[8];
};

// ---- TDM: stream a [tileCh x tilePix] f16 tile from planar [64][PTOT] ------
// D# per CDNA5 ISA 8.3/8.4: group0 = {count/type, lds_addr, global_addr},
// group1 = {data_size, tensor dims/strides, tile dims}. 2D tensor -> groups
// 2/3 zero. Issued by one wave; completion tracked with TENSORcnt.
__device__ __forceinline__ void tdm_load_tile_f16(const _Float16* gbase, int p0,
                                                  const _Float16* ldsTile,
                                                  int tilePix, int tileCh) {
    unsigned long long ga = (unsigned long long)gbase + (unsigned long long)p0 * 2ull;
    unsigned lds_off = (unsigned)(unsigned long long)ldsTile;  // LDS byte offset
    v4u g0 = {};
    g0[0] = 1u;                                         // count=1 (valid), load
    g0[1] = lds_off;                                    // lds_addr [63:32]
    g0[2] = (unsigned)ga;                               // global_addr [95:64]
    g0[3] = ((unsigned)(ga >> 32) & 0x01ffffffu)        // global_addr [120:96]
            | 0x80000000u;                              // type=2 ("image")
    v8i g1 = {};
    g1[0] = 0x00010000;                                 // data_size=1 (2 bytes)
    g1[1] = 0;                                          // tensor_dim0[15:0]=0
    g1[2] = (int)((PTOT >> 16) | (64u << 16));          // t_dim0 hi | t_dim1=64
    g1[3] = (int)((unsigned)tilePix << 16);             // tile_dim0 = pixels
    g1[4] = tileCh;                                     // tile_dim1 = channels
    g1[5] = PTOT;                                       // tensor_dim0_stride
    g1[6] = 0;
    g1[7] = 0;
    v4i z = {};
#if defined(__clang_major__) && (__clang_major__ >= 23)
    v8i g4 = {};
    __builtin_amdgcn_tensor_load_to_lds(g0, g1, z, z, g4, 0);
#else
    __builtin_amdgcn_tensor_load_to_lds(g0, g1, z, z, 0);
#endif
}

// ---- WMMA fragment builders (contiguous K order, used by both A and B) -----------------
// A 16x32 f16: lanes 0-15 M=lane, K base 0; lanes 16-31 M=lane-16, K base +16.
//   element j of the 16 maps to K = kbase + j.
__device__ __forceinline__ v16h load_a_frag(const _Float16* W, int rowStride,
                                            int mbase, int ks, int lane) {
    HFrag f;
    int M = mbase + (lane & 15);
    int kb = ks * 32 + ((lane >> 4) ? 16 : 0);
#pragma unroll
    for (int j = 0; j < 16; ++j) {
        int K = kb + j;
        f.h[j] = W[M * rowStride + K];
    }
    return f.v;
}

// B 32x16 f16: lane gives N=lane&15; VGPR i holds K = (lane<16?0:16)+2i, +1.
// Pixel-major tile [pix][ch] (rowStride even): each K pair is one b32 read.
__device__ __forceinline__ v16h load_b_frag(const _Float16* tile, int rowStride,
                                            int nbase, int kbase, int lane) {
    HFrag f;
    int pix = nbase + (lane & 15);
    int k0  = kbase + ((lane >> 4) ? 16 : 0);
    const _Float16* src = tile + pix * rowStride + k0;
#pragma unroll
    for (int i = 0; i < 8; ++i)
        f.d[i] = *(const unsigned int*)(src + 2 * i);
    return f.v;
}

// Channel-major tile [ch][pixW] (TDM-native layout): two d16 reads per pair.
__device__ __forceinline__ v16h load_b_frag_cm(const _Float16* tileCM, int pixW,
                                               int nbase, int kbase, int lane) {
    HFrag f;
    int pix = nbase + (lane & 15);
    int k0  = kbase + ((lane >> 4) ? 16 : 0);
#pragma unroll
    for (int i = 0; i < 8; ++i) {
        f.h[2 * i]     = tileCM[(k0 + 2 * i) * pixW + pix];
        f.h[2 * i + 1] = tileCM[(k0 + 2 * i + 1) * pixW + pix];
    }
    return f.v;
}

// ---- 1) per-pixel LayerNorm over 64 channels, f32 NCHW -> f16 planar -------
__global__ void ln_kernel(const float* __restrict__ xin,
                          const float* __restrict__ w,
                          const float* __restrict__ bsh,
                          _Float16* __restrict__ out) {
    int p = blockIdx.x * 256 + threadIdx.x;
    int b = p >> 16, n = p & (NPIX - 1);
    const float* base = xin + b * (DIMC * NPIX) + n;
    float vals[DIMC], s = 0.f, s2 = 0.f;
#pragma unroll
    for (int c = 0; c < DIMC; ++c) {
        float v = base[c << 16];
        vals[c] = v; s += v; s2 += v * v;
    }
    float mean = s * (1.f / 64.f);
    float var  = s2 * (1.f / 64.f) - mean * mean;
    float rs   = rsqrtf(var + 1e-5f);
#pragma unroll
    for (int c = 0; c < DIMC; ++c)
        out[c * PTOT + p] = (_Float16)((vals[c] - mean) * rs * w[c] + bsh[c]);
}

// ---- 2) conv1x1 as WMMA GEMM: out[MOxP] = W[MOx64] * in[64xP] --------------
// Input tile (64ch x 128px) DMA'd to LDS by the TDM while all waves convert
// the f32 weights to f16 in LDS.
template <int MO>
__global__ void conv1x1_f16_kernel(const _Float16* __restrict__ xin,
                                   const float* __restrict__ W,
                                   _Float16* __restrict__ out) {
    __shared__ _Float16 Wlds[MO * 64];
    __shared__ _Float16 tile[64 * 128];     // channel-major, TDM-filled
    int tid = threadIdx.x;
    int p0 = blockIdx.x * 128;
    if (tid < 32) tdm_load_tile_f16(xin, p0, tile, 128, 64);
    for (int i = tid; i < MO * 64; i += 256) Wlds[i] = (_Float16)W[i];
    if (tid < 32) __builtin_amdgcn_s_wait_tensorcnt(0);
    __syncthreads();
    int wv = tid >> 5, lane = tid & 31, hi = lane >> 4;
    const int NM = MO / 16;          // 4 or 8 m-fragments
    int mf = wv % NM;
    int nf0 = wv / NM;
    const int nstep = 8 / NM;        // 2 or 1
    v16h a0 = load_a_frag(Wlds, 64, mf * 16, 0, lane);
    v16h a1 = load_a_frag(Wlds, 64, mf * 16, 1, lane);
    for (int nf = nf0; nf < 8; nf += nstep) {
        v8f acc = {};
        v16h b0 = load_b_frag_cm(tile, 128, nf * 16, 0, lane);
        v16h b1 = load_b_frag_cm(tile, 128, nf * 16, 32, lane);
        acc = __builtin_amdgcn_wmma_f32_16x16x32_f16(false, a0, false, b0,
                                                     (short)0, acc, false, false);
        acc = __builtin_amdgcn_wmma_f32_16x16x32_f16(false, a1, false, b1,
                                                     (short)0, acc, false, false);
#pragma unroll
        for (int r = 0; r < 8; ++r) {
            int M = mf * 16 + r + (hi ? 8 : 0);
            int pix = p0 + nf * 16 + (lane & 15);
            out[M * PTOT + pix] = (_Float16)acc[r];
        }
    }
}

// ---- 3) depthwise 3x3, 64 channels per launch ------------------------------
__global__ void dwconv_kernel(const _Float16* __restrict__ src,
                              const float* __restrict__ Kw,
                              _Float16* __restrict__ dst) {
    int g = blockIdx.x * 256 + threadIdx.x;
    int c = g >> 18;                  // /PTOT
    int p = g & (PTOT - 1);
    int n = p & (NPIX - 1);
    int hh = n >> 8, ww = n & 255;
    const float* kc = Kw + c * 9;
    const _Float16* sp = src + c * PTOT + (p - n);   // (c, b) plane base
    float acc = 0.f;
#pragma unroll
    for (int dy = -1; dy <= 1; ++dy)
#pragma unroll
        for (int dx = -1; dx <= 1; ++dx) {
            int yy = hh + dy, xx = ww + dx;
            if ((unsigned)yy < 256u && (unsigned)xx < 256u)
                acc += kc[(dy + 1) * 3 + (dx + 1)] * (float)sp[(yy << 8) + xx];
        }
    dst[c * PTOT + p] = (_Float16)acc;
}

// ---- 4) per-(b,head): sum q^2, sum k^2, 8x8 Gram over N --------------------
// Gbuf layout per (b,h): [0..7]=qq  [8..15]=kk  [16..79]=gram[c][d]
__global__ void gram_kernel(const _Float16* __restrict__ q,
                            const _Float16* __restrict__ k,
                            float* __restrict__ Gbuf) {
    __shared__ float red[80];
    int bh = blockIdx.x >> 6;         // 0..31
    int chunk = blockIdx.x & 63;
    int b = bh >> 3, h = bh & 7;
    int ch0 = h * 8;
    int pbase = b * NPIX + chunk * 1024;
    float qq[8] = {}, kk[8] = {}, g[64] = {};
    for (int it = 0; it < 4; ++it) {
        int p = pbase + it * 256 + threadIdx.x;
        float qv[8], kv[8];
#pragma unroll
        for (int c = 0; c < 8; ++c) {
            qv[c] = (float)q[(ch0 + c) * PTOT + p];
            kv[c] = (float)k[(ch0 + c) * PTOT + p];
            qq[c] += qv[c] * qv[c];
            kk[c] += kv[c] * kv[c];
        }
#pragma unroll
        for (int c = 0; c < 8; ++c)
#pragma unroll
            for (int d = 0; d < 8; ++d) g[c * 8 + d] += qv[c] * kv[d];
    }
    if (threadIdx.x < 80) red[threadIdx.x] = 0.f;
    __syncthreads();
#pragma unroll
    for (int c = 0; c < 8; ++c) {
        atomicAdd(&red[c], qq[c]);
        atomicAdd(&red[8 + c], kk[c]);
    }
#pragma unroll
    for (int i = 0; i < 64; ++i) atomicAdd(&red[16 + i], g[i]);
    __syncthreads();
    if (threadIdx.x < 80)
        atomicAdd(&Gbuf[bh * 80 + threadIdx.x], red[threadIdx.x]);
}

// ---- 5) softmax(attn) then fold projection: M[b] = Wproj * blockdiag(attn) -
__global__ void attn_make_M(const float* __restrict__ Gbuf,
                            const float* __restrict__ temp,
                            const float* __restrict__ Wproj,
                            _Float16* __restrict__ Mbuf) {
    __shared__ float A[4 * 8 * 8 * 8];   // [b][h][c][d]
    int t = threadIdx.x;
    {
        int b = t >> 6, h = (t >> 3) & 7, c = t & 7;
        const float* gb = Gbuf + (b * 8 + h) * 80;
        float nq = fmaxf(sqrtf(gb[c]), 1e-12f);
        float raw[8], mx = -1e30f;
#pragma unroll
        for (int d = 0; d < 8; ++d) {
            float nk = fmaxf(sqrtf(gb[8 + d]), 1e-12f);
            raw[d] = gb[16 + c * 8 + d] / (nq * nk) * temp[h];
            mx = fmaxf(mx, raw[d]);
        }
        float s = 0.f;
#pragma unroll
        for (int d = 0; d < 8; ++d) { raw[d] = __expf(raw[d] - mx); s += raw[d]; }
        float inv = 1.f / s;
#pragma unroll
        for (int d = 0; d < 8; ++d)
            A[((b * 8 + h) * 8 + c) * 8 + d] = raw[d] * inv;
    }
    __syncthreads();
    {
        int b = t >> 6, o = t & 63;
        for (int j = 0; j < 64; ++j) {
            int h = j >> 3, d = j & 7;
            float s = 0.f;
#pragma unroll
            for (int cc = 0; cc < 8; ++cc)
                s += Wproj[o * 64 + h * 8 + cc] * A[((b * 8 + h) * 8 + cc) * 8 + d];
            Mbuf[(b * 64 + o) * 64 + j] = (_Float16)s;
        }
    }
}

// ---- 6) x1 = x + M[b] @ v  (WMMA, per-batch A matrix) ----------------------
__global__ void attnproj_kernel(const _Float16* __restrict__ vin,
                                const _Float16* __restrict__ Mbuf,
                                const float* __restrict__ x,
                                float* __restrict__ out) {
    __shared__ _Float16 Mlds[64 * 64];
    __shared__ _Float16 tile[128 * 68];
    int tid = threadIdx.x;
    int p0 = blockIdx.x * 128;
    int b = p0 >> 16;
    for (int i = tid; i < 64 * 64; i += 256) Mlds[i] = Mbuf[b * 4096 + i];
    for (int e = tid; e < 64 * 64; e += 256) {
        int ch = e >> 6, px2 = (e & 63) * 2;
        unsigned int w2 = *(const unsigned int*)(vin + ch * PTOT + p0 + px2);
        HFrag t; t.d[0] = w2;
        tile[px2 * 68 + ch]       = t.h[0];
        tile[(px2 + 1) * 68 + ch] = t.h[1];
    }
    __syncthreads();
    int wv = tid >> 5, lane = tid & 31, hi = lane >> 4;
    int mf = wv & 3;
    int nf0 = wv >> 2;
    v16h a0 = load_a_frag(Mlds, 64, mf * 16, 0, lane);
    v16h a1 = load_a_frag(Mlds, 64, mf * 16, 1, lane);
    for (int nf = nf0; nf < 8; nf += 2) {
        v8f acc = {};
        v16h b0 = load_b_frag(tile, 68, nf * 16, 0, lane);
        v16h b1 = load_b_frag(tile, 68, nf * 16, 32, lane);
        acc = __builtin_amdgcn_wmma_f32_16x16x32_f16(false, a0, false, b0,
                                                     (short)0, acc, false, false);
        acc = __builtin_amdgcn_wmma_f32_16x16x32_f16(false, a1, false, b1,
                                                     (short)0, acc, false, false);
#pragma unroll
        for (int r = 0; r < 8; ++r) {
            int c = mf * 16 + r + (hi ? 8 : 0);
            int pixg = p0 + nf * 16 + (lane & 15);
            int n = pixg & (NPIX - 1);
            int goff = ((b * 64 + c) << 16) + n;      // NCHW offset
            out[goff] = x[goff] + acc[r];
        }
    }
}

// ---- 8) fused FFN: W12 GEMM (row-permuted) -> SiLU gate -> W3 GEMM + resid -
// W12 rows permuted so (x1[j], x2[j]) land in accumulator pair (2t, 2t+1):
//   row' even -> W12[row'/2] (x1 half), row' odd -> W12[170 + row'/2] (x2 half)
// Input tile (64ch x 64px) arrives via TDM; gated activations h live in LDS
// pixel-major for the second GEMM.
__global__ void ffn_kernel(const _Float16* __restrict__ xn2,
                           const float* __restrict__ W12,
                           const float* __restrict__ W3,
                           float* __restrict__ out) {
    extern __shared__ _Float16 smem[];
    _Float16* W12l = smem;               // 352*64
    _Float16* tile = W12l + 352 * 64;    // 64ch * 64px channel-major (TDM)
    _Float16* hl   = tile + 64 * 64;     // 64px * 196 (gated activations)
    _Float16* W3l  = hl + 64 * 196;      // 64 * 192 (K padded 170->192)
    int tid = threadIdx.x;
    int p0 = blockIdx.x * 64;
    if (tid < 32) tdm_load_tile_f16(xn2, p0, tile, 64, 64);
    for (int i = tid; i < 352 * 64; i += 256) {
        int mp = i >> 6, kk = i & 63;
        int half = mp & 1, r = mp >> 1;
        float v = (r < 170) ? W12[(half ? (170 + r) : r) * 64 + kk] : 0.f;
        W12l[i] = (_Float16)v;
    }
    for (int i = tid; i < 64 * 192; i += 256) {
        int o = i / 192, kk = i % 192;
        W3l[i] = (_Float16)((kk < 170) ? W3[o * 170 + kk] : 0.f);
    }
    for (int i = tid; i < 64 * 196; i += 256) hl[i] = (_Float16)0.f;
    if (tid < 32) __builtin_amdgcn_s_wait_tensorcnt(0);
    __syncthreads();
    int wv = tid >> 5, lane = tid & 31, hi = lane >> 4;
    // GEMM1: 22 mfrags (352 rows) x 4 nfrags, K=64
    for (int f = wv; f < 88; f += 8) {
        int mf = f >> 2, nf = f & 3;
        v8f acc = {};
        v16h a0 = load_a_frag(W12l, 64, mf * 16, 0, lane);
        v16h b0 = load_b_frag_cm(tile, 64, nf * 16, 0, lane);
        acc = __builtin_amdgcn_wmma_f32_16x16x32_f16(false, a0, false, b0,
                                                     (short)0, acc, false, false);
        v16h a1 = load_a_frag(W12l, 64, mf * 16, 1, lane);
        v16h b1 = load_b_frag_cm(tile, 64, nf * 16, 32, lane);
        acc = __builtin_amdgcn_wmma_f32_16x16x32_f16(false, a1, false, b1,
                                                     (short)0, acc, false, false);
        int pix = nf * 16 + (lane & 15);
#pragma unroll
        for (int t = 0; t < 4; ++t) {
            float x1v = acc[2 * t], x2v = acc[2 * t + 1];
            float gg = (x1v / (1.f + __expf(-x1v))) * x2v;   // silu(x1)*x2
            int hrow = mf * 8 + t + (hi ? 4 : 0);             // 0..175
            hl[pix * 196 + hrow] = (_Float16)gg;
        }
    }
    __syncthreads();
    // GEMM2: M=64, K=192 (6 k-steps), 4x4 frags; residual add in-place
    int b_ = p0 >> 16;
    for (int f = wv; f < 16; f += 8) {
        int mf = f >> 2, nf = f & 3;
        v8f acc = {};
#pragma unroll
        for (int ks = 0; ks < 6; ++ks) {
            v16h a = load_a_frag(W3l, 192, mf * 16, ks, lane);
            v16h b = load_b_frag(hl, 196, nf * 16, ks * 32, lane);
            acc = __builtin_amdgcn_wmma_f32_16x16x32_f16(false, a, false, b,
                                                         (short)0, acc, false, false);
        }
#pragma unroll
        for (int r = 0; r < 8; ++r) {
            int c = mf * 16 + r + (hi ? 8 : 0);
            int pixg = p0 + nf * 16 + (lane & 15);
            int n = pixg & (NPIX - 1);
            int goff = ((b_ * 64 + c) << 16) + n;
            out[goff] = out[goff] + acc[r];
        }
    }
}

__global__ void zero_kernel(float* __restrict__ p, int n) {
    int i = blockIdx.x * 256 + threadIdx.x;
    if (i < n) p[i] = 0.f;
}

// ---------------------------------------------------------------------------
extern "C" void kernel_launch(void* const* d_in, const int* in_sizes, int n_in,
                              void* d_out, int out_size, void* d_ws, size_t ws_size,
                              hipStream_t stream) {
    (void)in_sizes; (void)n_in; (void)out_size; (void)ws_size;
    const float* x      = (const float*)d_in[0];
    const float* y      = (const float*)d_in[1];
    const float* ln1x_w = (const float*)d_in[2];
    const float* ln1x_b = (const float*)d_in[3];
    const float* ln1y_w = (const float*)d_in[4];
    const float* ln1y_b = (const float*)d_in[5];
    const float* ln2_w  = (const float*)d_in[6];
    const float* ln2_b  = (const float*)d_in[7];
    const float* Wq     = (const float*)d_in[8];
    const float* Wq_dw  = (const float*)d_in[9];
    const float* Wkv    = (const float*)d_in[10];
    const float* Wkv_dw = (const float*)d_in[11];
    const float* Wproj  = (const float*)d_in[12];
    const float* temp   = (const float*)d_in[13];
    const float* W12    = (const float*)d_in[14];
    const float* W3     = (const float*)d_in[15];
    float* outp = (float*)d_out;

    // workspace layout (bytes); regions aliased across phases. ~192 MB total.
    char* ws = (char*)d_ws;
    const size_t SZ64 = (size_t)DIMC * PTOT * sizeof(_Float16);  // 33554432
    _Float16* bufA = (_Float16*)(ws);                    // xn   -> q_dw
    _Float16* bufB = (_Float16*)(ws + SZ64);             // yn   -> k_dw
    _Float16* bufC = (_Float16*)(ws + 2 * SZ64);         // q_pre-> v_dw
    _Float16* bufD = (_Float16*)(ws + 3 * SZ64);         // kv_pre (128 ch)
    _Float16* bufDv = (_Float16*)(ws + 4 * SZ64);        // v half of kv_pre
    _Float16* bufE = (_Float16*)(ws + 5 * SZ64);         // xn2
    float*    Gbuf = (float*)(ws + 6 * SZ64);            // 32*80 f32
    _Float16* Mbuf = (_Float16*)(ws + 6 * SZ64 + 10240); // 4*64*64 f16

    zero_kernel<<<10, 256, 0, stream>>>(Gbuf, 32 * 80);

    ln_kernel<<<PTOT / 256, 256, 0, stream>>>(x, ln1x_w, ln1x_b, bufA);
    ln_kernel<<<PTOT / 256, 256, 0, stream>>>(y, ln1y_w, ln1y_b, bufB);

    conv1x1_f16_kernel<64><<<PTOT / 128, 256, 0, stream>>>(bufA, Wq, bufC);
    conv1x1_f16_kernel<128><<<PTOT / 128, 256, 0, stream>>>(bufB, Wkv, bufD);

    dwconv_kernel<<<(64 * PTOT) / 256, 256, 0, stream>>>(bufC, Wq_dw, bufA);       // q
    dwconv_kernel<<<(64 * PTOT) / 256, 256, 0, stream>>>(bufD, Wkv_dw, bufB);      // k
    dwconv_kernel<<<(64 * PTOT) / 256, 256, 0, stream>>>(bufDv, Wkv_dw + 64 * 9, bufC); // v

    gram_kernel<<<32 * 64, 256, 0, stream>>>(bufA, bufB, Gbuf);
    attn_make_M<<<1, 256, 0, stream>>>(Gbuf, temp, Wproj, Mbuf);

    attnproj_kernel<<<PTOT / 128, 256, 0, stream>>>(bufC, Mbuf, x, outp); // x1 -> d_out

    ln_kernel<<<PTOT / 256, 256, 0, stream>>>(outp, ln2_w, ln2_b, bufE);

    size_t ffn_lds = (size_t)(352 * 64 + 64 * 64 + 64 * 196 + 64 * 192) * sizeof(_Float16);
    ffn_kernel<<<PTOT / 64, 256, ffn_lds, stream>>>(bufE, W12, W3, outp);
}
